// Seq2Seq_2422361555569
// MI455X (gfx1250) — matrix operand order repaired and checked by reading, not compiled, not run on hardware
//
#include <hip/hip_runtime.h>
#include <stdint.h>

typedef __attribute__((ext_vector_type(16))) _Float16 v16h;
typedef __attribute__((ext_vector_type(8)))  _Float16 v8h;
typedef __attribute__((ext_vector_type(4)))  _Float16 v4h;
typedef __attribute__((ext_vector_type(8)))  float    v8f;
typedef __attribute__((ext_vector_type(4)))  float    v4f;

#define B_DIM 4096
#define H_DIM 2048
#define BM 128
#define BN 128
#define BK 32
#define KSTEPS (H_DIM / BK)
#define NBLK ((B_DIM / BM) * (B_DIM / BN))   // 1024 blocks -> 2 floats each

// ---------------------------------------------------------------------------
// Shared epilogue: sign flip + stable log-sigmoid + deterministic reduction.
// C/D layout (ISA 7.12.2): element j of acc[t] is S[m][n] with
//   m = bm + wave*16 + j + hi*8 ,  n = bn + t*16 + l16
// ---------------------------------------------------------------------------
__device__ __forceinline__ void epilogue_reduce(
    const v8f* acc, int bm, int bn, int wave, int l16, int hi,
    float* redA, float* redD, float* __restrict__ partials)
{
    float sum_all  = 0.0f;
    float sum_diag = 0.0f;
    #pragma unroll
    for (int t = 0; t < 8; ++t) {
        const int n = bn + t * 16 + l16;
        #pragma unroll
        for (int j = 0; j < 8; ++j) {
            const int m = bm + wave * 16 + j + hi * 8;
            const float s  = acc[t][j];
            const float x  = (m == n) ? s : -s;
            const float ll = fminf(x, 0.0f) - log1pf(expf(-fabsf(x)));
            sum_all += ll;
            if (m == n) sum_diag += ll;
        }
    }
    #pragma unroll
    for (int off = 16; off > 0; off >>= 1) {
        sum_all  += __shfl_down(sum_all,  off, 32);
        sum_diag += __shfl_down(sum_diag, off, 32);
    }
    const int lane = (hi << 4) | l16;
    if (lane == 0) { redA[wave] = sum_all; redD[wave] = sum_diag; }
    __syncthreads();
    if (threadIdx.x == 0) {
        float a = 0.0f, d = 0.0f;
        #pragma unroll
        for (int w = 0; w < 8; ++w) { a += redA[w]; d += redD[w]; }
        const int bid = blockIdx.y * gridDim.x + blockIdx.x;
        partials[2 * bid + 0] = a;
        partials[2 * bid + 1] = d;
    }
}

// ---------------------------------------------------------------------------
// Inner product step: A fragment + 8 WMMAs with B fragments software-
// pipelined one tile ahead so each ds_load pair hides under a full WMMA.
// ---------------------------------------------------------------------------
__device__ __forceinline__ void mma_strip(
    const _Float16* __restrict__ Ab, const _Float16* __restrict__ Bb,
    int arow, int kb, int l16, int hi, v8f* acc)
{
    const v8h alo = *(const v8h*)(Ab + arow * BK + kb * 2);
    const v8h ahh = *(const v8h*)(Ab + arow * BK + kb * 2 + 8);
    v16h af;
    #pragma unroll
    for (int q = 0; q < 8; ++q) { af[q] = alo[q]; af[q + 8] = ahh[q]; }

    const _Float16* bbase = Bb + l16 * BK + hi * 16;
    v16h bf = *(const v16h*)(bbase);
    #pragma unroll
    for (int t = 0; t < 7; ++t) {
        const v16h bfn = *(const v16h*)(bbase + (t + 1) * (16 * BK));
        acc[t] = __builtin_amdgcn_wmma_f32_16x16x32_f16(
                     false, af, false, bf, (short)0, acc[t], false, false);
        bf = bfn;
    }
    acc[7] = __builtin_amdgcn_wmma_f32_16x16x32_f16(
                 false, af, false, bf, (short)0, acc[7], false, false);
}

// ---------------------------------------------------------------------------
// Kernel 1 (fast path): D-axis reduce + fp32 -> f16 convert, one pass.
// Non-temporal loads: the fp32 stream is dead after this; keep L2 for f16.
// ---------------------------------------------------------------------------
__global__ __launch_bounds__(256) void reduce_convert(
    const float* __restrict__ in,     // [B, 2, H] fp32
    _Float16* __restrict__ out)       // [B, H]    f16
{
    const int idx = (blockIdx.x * 256 + threadIdx.x) * 8;  // < B*H = 8.4M
    const int row = idx / H_DIM;
    const int col = idx % H_DIM;                            // H%8==0: no row cross
    const v4f* p0 = (const v4f*)(in + (row * 2 + 0) * H_DIM + col);
    const v4f* p1 = (const v4f*)(in + (row * 2 + 1) * H_DIM + col);
    const v4f a0 = __builtin_nontemporal_load(p0);
    const v4f a1 = __builtin_nontemporal_load(p0 + 1);
    const v4f b0 = __builtin_nontemporal_load(p1);
    const v4f b1 = __builtin_nontemporal_load(p1 + 1);
    v8h h;
    #pragma unroll
    for (int i = 0; i < 4; ++i) {
        h[i]     = (_Float16)(a0[i] + b0[i]);
        h[i + 4] = (_Float16)(a1[i] + b1[i]);
    }
    *(v8h*)(out + idx) = h;
}

// ---------------------------------------------------------------------------
// Kernel 2 (fast path): f16 WMMA GEMM with double-buffered async LDS staging.
// ---------------------------------------------------------------------------
__global__ __launch_bounds__(256) void gemm_logsig_f16(
    const _Float16* __restrict__ eh,  // [B, H] f16
    const _Float16* __restrict__ dh,  // [B, H] f16
    float* __restrict__ partials)
{
    __shared__ __align__(32) _Float16 At[2][BM * BK];   // contiguous: +8 KB per buf
    __shared__ __align__(32) _Float16 Bt[2][BN * BK];
    __shared__ float redA[8];
    __shared__ float redD[8];

    const int tid  = threadIdx.x;
    const int lane = tid & 31;
    const int wave = tid >> 5;
    const int l16  = lane & 15;
    const int hi   = lane >> 4;

    const int bm = blockIdx.y * BM;
    const int bn = blockIdx.x * BN;

    // Async staging tasks: 128 rows x 64 B per tile; task = (row, 16B segment).
    // Thread t owns tasks t and t+256.
    const int t0 = tid,       r0 = t0 >> 2, s0 = (t0 & 3) * 8;   // halves
    const int t1 = tid + 256, r1 = t1 >> 2, s1 = (t1 & 3) * 8;

    const uint64_t eBase = (uint64_t)(uintptr_t)eh;
    const uint64_t dBase = (uint64_t)(uintptr_t)dh;
    const uint64_t gA0 = eBase + 2ull * ((uint64_t)(bm + r0) * H_DIM + (uint64_t)s0);
    const uint64_t gA1 = eBase + 2ull * ((uint64_t)(bm + r1) * H_DIM + (uint64_t)s1);
    const uint64_t gB0 = dBase + 2ull * ((uint64_t)(bn + r0) * H_DIM + (uint64_t)s0);
    const uint64_t gB1 = dBase + 2ull * ((uint64_t)(bn + r1) * H_DIM + (uint64_t)s1);

    // LDS byte addresses in buffer 0; buffer b is +(b << 13).
    const uint32_t lA0 = (uint32_t)(uintptr_t)&At[0][r0 * BK + s0];
    const uint32_t lA1 = (uint32_t)(uintptr_t)&At[0][r1 * BK + s1];
    const uint32_t lB0 = (uint32_t)(uintptr_t)&Bt[0][r0 * BK + s0];
    const uint32_t lB1 = (uint32_t)(uintptr_t)&Bt[0][r1 * BK + s1];

    auto stage = [&](uint32_t bufoff, uint64_t goff) {
        asm volatile("global_load_async_to_lds_b128 %0, %1, off"
                     :: "v"(lA0 + bufoff), "v"(gA0 + goff) : "memory");
        asm volatile("global_load_async_to_lds_b128 %0, %1, off"
                     :: "v"(lA1 + bufoff), "v"(gA1 + goff) : "memory");
        asm volatile("global_load_async_to_lds_b128 %0, %1, off"
                     :: "v"(lB0 + bufoff), "v"(gB0 + goff) : "memory");
        asm volatile("global_load_async_to_lds_b128 %0, %1, off"
                     :: "v"(lB1 + bufoff), "v"(gB1 + goff) : "memory");
    };

    v8f acc[8] = {};

    stage(0, 0);
    asm volatile("s_wait_asynccnt 0" ::: "memory");
    __syncthreads();

    const int arow = wave * 16 + l16;
    const int kb   = hi * 8;

    #pragma unroll 1
    for (int i = 0; i < KSTEPS; ++i) {
        const int cur = i & 1;
        if (i + 1 < KSTEPS)
            stage((uint32_t)((cur ^ 1) << 13), (uint64_t)(i + 1) * (BK * 2));

        const _Float16* Ab = At[0] + cur * (BM * BK);
        const _Float16* Bb = Bt[0] + cur * (BN * BK);

        mma_strip(Ab, Bb, arow, kb, l16, hi, acc);

        asm volatile("s_wait_asynccnt 0" ::: "memory");
        __syncthreads();
    }

    epilogue_reduce(acc, bm, bn, wave, l16, hi, redA, redD, partials);
}

// ---------------------------------------------------------------------------
// Fallback (small ws): fused convert-in-kernel version.
// ---------------------------------------------------------------------------
__global__ __launch_bounds__(256) void sim_gemm_logsig(
    const float* __restrict__ enc, const float* __restrict__ dec,
    float* __restrict__ partials)
{
    __shared__ __align__(32) _Float16 At[BM * BK];
    __shared__ __align__(32) _Float16 Bt[BN * BK];
    __shared__ float redA[8];
    __shared__ float redD[8];

    const int tid  = threadIdx.x;
    const int lane = tid & 31;
    const int wave = tid >> 5;
    const int l16  = lane & 15;
    const int hi   = lane >> 4;

    const int bm = blockIdx.y * BM;
    const int bn = blockIdx.x * BN;

    v8f acc[8] = {};

    const int srow = tid >> 3;
    const int skk  = (tid & 7) * 4;

    #pragma unroll 1
    for (int k0 = 0; k0 < H_DIM; k0 += BK) {
        #pragma unroll
        for (int rr = 0; rr < 4; ++rr) {
            const int r = rr * 32 + srow;
            {
                const v4f a0 = *(const v4f*)(enc + ((bm + r) * 2 + 0) * H_DIM + k0 + skk);
                const v4f a1 = *(const v4f*)(enc + ((bm + r) * 2 + 1) * H_DIM + k0 + skk);
                v4h h;
                #pragma unroll
                for (int q = 0; q < 4; ++q) h[q] = (_Float16)(a0[q] + a1[q]);
                *(v4h*)(At + r * BK + skk) = h;
            }
            {
                const v4f b0 = *(const v4f*)(dec + ((bn + r) * 2 + 0) * H_DIM + k0 + skk);
                const v4f b1 = *(const v4f*)(dec + ((bn + r) * 2 + 1) * H_DIM + k0 + skk);
                v4h h;
                #pragma unroll
                for (int q = 0; q < 4; ++q) h[q] = (_Float16)(b0[q] + b1[q]);
                *(v4h*)(Bt + r * BK + skk) = h;
            }
        }
        __syncthreads();

        mma_strip(At, Bt, wave * 16 + l16, hi * 8, l16, hi, acc);

        __syncthreads();
    }

    epilogue_reduce(acc, bm, bn, wave, l16, hi, redA, redD, partials);
}

// ---------------------------------------------------------------------------
// Final reduction of per-block partials to the two output scalars.
// ---------------------------------------------------------------------------
__global__ __launch_bounds__(256) void finalize_loss(
    const float* __restrict__ partials, float* __restrict__ out)
{
    __shared__ float redA[8];
    __shared__ float redD[8];
    float a = 0.0f, d = 0.0f;
    for (int i = threadIdx.x; i < NBLK; i += 256) {
        a += partials[2 * i + 0];
        d += partials[2 * i + 1];
    }
    #pragma unroll
    for (int off = 16; off > 0; off >>= 1) {
        a += __shfl_down(a, off, 32);
        d += __shfl_down(d, off, 32);
    }
    const int lane = threadIdx.x & 31, w = threadIdx.x >> 5;
    if (lane == 0) { redA[w] = a; redD[w] = d; }
    __syncthreads();
    if (threadIdx.x == 0) {
        float A = 0.0f, D = 0.0f;
        #pragma unroll
        for (int i = 0; i < 8; ++i) { A += redA[i]; D += redD[i]; }
        const float total = A + 39.0f * D;   // sum(LL) + 39*sum(diag)
        out[0] = -total / (float)B_DIM;           // logLoss
        out[1] = -(40.0f * D) / (float)B_DIM;     // diagonalLoss
    }
}

extern "C" void kernel_launch(void* const* d_in, const int* in_sizes, int n_in,
                              void* d_out, int out_size, void* d_ws, size_t ws_size,
                              hipStream_t stream) {
    (void)in_sizes; (void)n_in; (void)out_size;
    const float* enc = (const float*)d_in[0];
    const float* dec = (const float*)d_in[1];
    float* out = (float*)d_out;

    const size_t EH_ELT   = (size_t)B_DIM * H_DIM;        // 8.39M halves
    const size_t EH_BYTES = EH_ELT * 2;                   // 16 MB
    const size_t NEED     = 2 * EH_BYTES + (size_t)NBLK * 2 * sizeof(float);

    dim3 grid(B_DIM / BN, B_DIM / BM);                    // 32 x 32

    if (ws_size >= NEED) {
        _Float16* eh = (_Float16*)d_ws;
        _Float16* dh = eh + EH_ELT;
        float* partials = (float*)((char*)d_ws + 2 * EH_BYTES);
        const int cvt_blocks = (int)(EH_ELT / (256 * 8)); // 4096
        reduce_convert<<<cvt_blocks, 256, 0, stream>>>(enc, eh);
        reduce_convert<<<cvt_blocks, 256, 0, stream>>>(dec, dh);
        gemm_logsig_f16<<<grid, 256, 0, stream>>>(eh, dh, partials);
        finalize_loss<<<1, 256, 0, stream>>>(partials, out);
    } else {
        float* partials = (float*)d_ws;                   // 8 KB
        sim_gemm_logsig<<<grid, 256, 0, stream>>>(enc, dec, partials);
        finalize_loss<<<1, 256, 0, stream>>>(partials, out);
    }
}
